// EGNNLayer_66649302499539
// MI455X (gfx1250) — matrix operand order repaired and checked by reading, not compiled, not run on hardware
//
#include <hip/hip_runtime.h>
#include <hip/hip_bf16.h>
#include <math.h>

#define B_ 4
#define N_ 128
#define P_ 512
#define H_ 64
#define E_ 32
#define O_ 64
#define NJ_ (N_ + P_)   // 640 neighbors per node
#define KF_ 192         // padded edge-feature dim (169 real)

// output layout (flat f32): upd_q | upd_x | upd_torsions | o
#define UQ_OFF  0
#define UX_OFF  2048
#define TOR_OFF 3584
#define OO_OFF  10752

// f16 weight blob (fragment-major), in halves:
//   Wm1: 24 frags (6 kb x 4 nt) -> 24*32*16 = 12288
//   Wm2:  8 frags (2 kb x 4 nt) ->  4096   @ 12288
//   Wx1:  8 frags               ->  4096   @ 16384
#define WSZ_HALVES 20480
#define WSZ_BYTES  (WSZ_HALVES * 2)     // 40960 B = 5120 x 8B
// workspace layout (bytes)
#define WS_MSUM_OFF  40960              // f32 msum[B*N][64]
#define WS_DX_OFF    (40960 + 131072)   // f32 dxsum[B*N][3]

typedef __attribute__((ext_vector_type(16))) _Float16 v16h;
typedef __attribute__((ext_vector_type(8)))  _Float16 v8h;
typedef __attribute__((ext_vector_type(8)))  float    v8f;
typedef __attribute__((ext_vector_type(4)))  unsigned int u32x4;
typedef __attribute__((ext_vector_type(8)))  int i32x8;
typedef __attribute__((ext_vector_type(4)))  int i32x4;

__device__ __forceinline__ void lds_fence() {
  // per-wave LDS ordering point (LDS is in-order per wave; also a compiler
  // barrier so ds ops are not reordered across phases)
  __asm__ volatile("s_wait_dscnt 0" ::: "memory");
}

__device__ __forceinline__ v8f wmma_f16(v16h a, v16h b, v8f c) {
  return __builtin_amdgcn_wmma_f32_16x16x32_f16(false, a, false, b, (short)0, c,
                                                false, false);
}

// fragment-major load: fragment = 32 lanes x 16 halves, contiguous per lane
__device__ __forceinline__ v16h load_frag32(const _Float16* base, int fragIdx, int lane) {
  return *(const v16h*)(base + (fragIdx * 32 + lane) * 16);
}

// A-fragment element address: (row, k) -> frag kb=k>>5, lane=row+16*((k>>3)&1),
// idx = (k&7) | ((k>>4)&1)<<3  (8 consecutive aligned K -> 16 contiguous bytes)
__device__ __forceinline__ void store_afrag_elem(_Float16* base, int row, int k, float v) {
  const int kb   = k >> 5, kin = k & 31;
  const int ln   = row + ((kin >> 3) & 1) * 16;
  const int idx  = (kin & 7) | (((kin >> 4) & 1) << 3);
  base[(kb * 32 + ln) * 16 + idx] = (_Float16)v;
}

// store one aligned octet (8 halves) of a row into A-fragment layout
__device__ __forceinline__ void store_octet(_Float16* base, int row, int o, v8h pk) {
  const int k0  = o * 8;
  const int kb  = k0 >> 5, kin = k0 & 31;
  const int ln  = row + ((kin >> 3) & 1) * 16;
  const int idx = ((kin >> 4) & 1) * 8;
  *(v8h*)(base + (kb * 32 + ln) * 16 + idx) = pk;
}

__device__ __forceinline__ v8h pack8(const float4 a, const float4 b) {
  v8h p;
  p[0] = (_Float16)a.x; p[1] = (_Float16)a.y; p[2] = (_Float16)a.z; p[3] = (_Float16)a.w;
  p[4] = (_Float16)b.x; p[5] = (_Float16)b.y; p[6] = (_Float16)b.z; p[7] = (_Float16)b.w;
  return p;
}

// ------------- prep: fp32 weights -> f16 in B-fragment-major order ----------
// B-frag mapping: frag (kb,nt); lane L -> col = nt*16 + (L&15); within-lane
// half idx -> K = kb*32 + (L>>4)*8 + (idx&7) + (idx>>3)*16
__global__ void prep_kernel(const float* __restrict__ Wm1,
                            const float* __restrict__ Wm2,
                            const float* __restrict__ Wx1,
                            _Float16* __restrict__ w16) {
  const int t = blockIdx.x * 256 + threadIdx.x;
  if (t >= WSZ_HALVES) return;
  int rel, f;
  const float* W;
  int kmax;
  if (t < 12288)      { rel = t;          W = Wm1; kmax = 169; }
  else if (t < 16384) { rel = t - 12288;  W = Wm2; kmax = 64;  }
  else                { rel = t - 16384;  W = Wx1; kmax = 64;  }
  const int idx  = rel & 15;
  const int lane = (rel >> 4) & 31;
  f              = rel >> 9;
  const int kb = f >> 2, nt = f & 3;
  const int k   = kb * 32 + (lane >> 4) * 8 + (idx & 7) + (idx >> 3) * 16;
  const int col = nt * 16 + (lane & 15);
  w16[t] = (k < kmax) ? (_Float16)W[k * 64 + col] : (_Float16)0.0f;
}

// ---------------- edge kernel: one block (2 waves) per (b,i) -----------------
__global__ __launch_bounds__(64) void edge_kernel(
    const float* __restrict__ qG,  const float* __restrict__ xG,
    const float* __restrict__ hG,  const float* __restrict__ eG,
    const unsigned char* __restrict__ nmG,
    const float* __restrict__ pqG, const float* __restrict__ pxG,
    const float* __restrict__ peG, const float* __restrict__ phG,
    const unsigned char* __restrict__ pmG,
    const float* __restrict__ bm1G, const float* __restrict__ bm2G,
    const float* __restrict__ bx1G, const float* __restrict__ Wx2G,
    const float* __restrict__ bx2G,
    const _Float16* __restrict__ w16,
    float* __restrict__ msumG, float* __restrict__ dxG)
{
  __shared__ _Float16 sW[WSZ_HALVES];       // Wm1 frags @0, Wm2 @12288, Wx1 @16384
  __shared__ _Float16 sScr[2][6 * 32 * 16]; // per-wave: feat A-frags; reused for hid/m/t
  __shared__ float sMask[2][16];
  __shared__ float sMsum[64];
  __shared__ float sDx[3];
  __shared__ float sBm1[64], sBm2[64], sBx1[64], sWx2[64 * 3], sBx2[3];

  const int tid  = threadIdx.x;
  const int lane = tid & 31;
  const int wav  = tid >> 5;
  const int node = blockIdx.x;      // b*N + i
  const int b    = node >> 7;
  const int i    = node & (N_ - 1);

  // ---- stage f16 weight blob into LDS ----
#if __has_builtin(__builtin_amdgcn_tensor_load_to_lds)
  if (wav == 0) {
    // Tensor Data Mover: one 1-row tile of 5120 x 8B = 40960 B, global -> LDS.
    const unsigned lds_off = (unsigned)(size_t)(const void*)&sW[0]; // LDS aperture: addr[31:0] = LDS byte offset
    const unsigned long long ga = (unsigned long long)(size_t)w16;
    u32x4 g0;
    g0[0] = 1u;                                   // count=1, user descriptor
    g0[1] = lds_off;                              // lds_addr
    g0[2] = (unsigned)(ga & 0xffffffffu);         // global_addr[31:0]
    g0[3] = (unsigned)((ga >> 32) & 0x01ffffffu)  // global_addr[56:32]
            | (2u << 30);                         // type = 2 ("image")
    i32x8 g1;
    g1[0] = (int)(3u << 16);                      // data_size = 3 (8 bytes)
    g1[1] = (int)((WSZ_BYTES / 8) << 16);         // tensor_dim0[15:0] @ bits[63:48]
    g1[2] = (int)(1u << 16);                      // tensor_dim1 = 1
    g1[3] = (int)((WSZ_BYTES / 8) << 16);         // tile_dim0 @ bits[127:112]
    g1[4] = 1;                                    // tile_dim1 = 1, tile_dim2 = 0
    g1[5] = WSZ_BYTES / 8;                        // tensor_dim0_stride[31:0]
    g1[6] = 0; g1[7] = 0;
    const i32x4 gz4 = {0, 0, 0, 0};
    const i32x8 gz8 = {0, 0, 0, 0, 0, 0, 0, 0};
    // 6-arg form (clang-23 / therock-10.0 headers)
    __builtin_amdgcn_tensor_load_to_lds(g0, g1, gz4, gz4, gz8, 0);
    __builtin_amdgcn_s_wait_tensorcnt(0);
  }
#else
  {
    const uint4* src = (const uint4*)w16;
    uint4* dst = (uint4*)sW;
    for (int t = tid; t < WSZ_HALVES / 8; t += 64) dst[t] = src[t];
  }
#endif
  if (tid < 64) { sBm1[tid] = bm1G[tid]; sBm2[tid] = bm2G[tid];
                  sBx1[tid] = bx1G[tid]; sMsum[tid] = 0.0f; }
  for (int t = tid; t < 192; t += 64) sWx2[t] = Wx2G[t];
  if (tid < 3) { sBx2[tid] = bx2G[tid]; sDx[tid] = 0.0f; }
  __syncthreads();

  _Float16* sF  = sScr[wav];
  float*    sMk = sMask[wav];
  const int e16   = lane & 15;
  const int hs    = lane >> 4;
  const int col16 = lane & 15;

  // loop-invariant per-lane bias values (one per N-tile)
  float bM1[4], bM2[4], bX1[4];
#pragma unroll
  for (int nt = 0; nt < 4; ++nt) {
    bM1[nt] = sBm1[nt * 16 + col16];
    bM2[nt] = sBm2[nt * 16 + col16];
    bX1[nt] = sBx1[nt * 16 + col16];
  }

  const float xi0 = xG[node * 3 + 0], xi1 = xG[node * 3 + 1], xi2 = xG[node * 3 + 2];
  const float qiw = qG[node * 4 + 0], qix = qG[node * 4 + 1];
  const float qiy = qG[node * 4 + 2], qiz = qG[node * 4 + 3];
  const bool  mi  = nmG[node] != 0;
  const float* hip = hG + node * H_;

  for (int tile = wav * 20; tile < wav * 20 + 20; ++tile) {
    const int jbase = tile * 16;
    const int j     = jbase + e16;

    const float* hjp = (j < N_) ? (hG + (b * N_ + j) * H_)
                                : (phG + (b * P_ + (j - N_)) * H_);
    const float* evp = (j < N_) ? (eG  + ((b * N_ + i) * N_ + j) * E_)
                                : (peG + ((b * N_ + i) * P_ + (j - N_)) * E_);
    // per-edge geometry sources (both half-waves load; lane<16 keeps for dx)
    float xj0, xj1, xj2, qjw, qjx, qjy, qjz; bool mj;
    if (j < N_) {
      xj0 = xG[(b * N_ + j) * 3 + 0]; xj1 = xG[(b * N_ + j) * 3 + 1]; xj2 = xG[(b * N_ + j) * 3 + 2];
      qjw = qG[(b * N_ + j) * 4 + 0]; qjx = qG[(b * N_ + j) * 4 + 1];
      qjy = qG[(b * N_ + j) * 4 + 2]; qjz = qG[(b * N_ + j) * 4 + 3];
      mj = (nmG[b * N_ + j] != 0) && (j != i);
    } else {
      const int jp = j - N_;
      xj0 = pxG[(b * P_ + jp) * 3 + 0]; xj1 = pxG[(b * P_ + jp) * 3 + 1]; xj2 = pxG[(b * P_ + jp) * 3 + 2];
      qjw = pqG[(b * P_ + jp) * 4 + 0]; qjx = pqG[(b * P_ + jp) * 4 + 1];
      qjy = pqG[(b * P_ + jp) * 4 + 2]; qjz = pqG[(b * P_ + jp) * 4 + 3];
      mj = pmG[b * P_ + jp] != 0;
    }
    const float maskv = (mi && mj) ? 1.0f : 0.0f;

    // ---- build 16x192 feature tile directly in A-fragment layout ----
    if (lane < 16) {
      sMk[e16] = maskv;
#pragma unroll
      for (int o = 0; o < 8; ++o) {   // k 0..63 : h_i
        const float4 a = *(const float4*)(hip + o * 8);
        const float4 c = *(const float4*)(hip + o * 8 + 4);
        store_octet(sF, e16, o, pack8(a, c));
      }
#pragma unroll
      for (int o = 8; o < 12; ++o) {  // k 64..95 : h_j[0..31]
        const float4 a = *(const float4*)(hjp + (o - 8) * 8);
        const float4 c = *(const float4*)(hjp + (o - 8) * 8 + 4);
        store_octet(sF, e16, o, pack8(a, c));
      }
    } else {
#pragma unroll
      for (int o = 12; o < 16; ++o) { // k 96..127 : h_j[32..63]
        const float4 a = *(const float4*)(hjp + (o - 12) * 8 + 32);
        const float4 c = *(const float4*)(hjp + (o - 12) * 8 + 36);
        store_octet(sF, e16, o, pack8(a, c));
      }
#pragma unroll
      for (int o = 16; o < 20; ++o) { // k 128..159 : e_ij
        const float4 a = *(const float4*)(evp + (o - 16) * 8);
        const float4 c = *(const float4*)(evp + (o - 16) * 8 + 4);
        store_octet(sF, e16, o, pack8(a, c));
      }
      // geometry -> k 160..167 ; qdot + zero pad -> k 168..191
      const float d0 = xi0 - xj0, d1 = xi1 - xj1, d2c = xi2 - xj2;
      const float dd = d0 * d0 + d1 * d1 + d2c * d2c;
      const float qd = fabsf(qiw * qjw + qix * qjx + qiy * qjy + qiz * qjz);
      const float cw = qjw, cx = -qjx, cy = -qjy, cz = -qjz;  // conj(q_j)
      const float t0 = 2.f * (cy * d2c - cz * d1);
      const float t1 = 2.f * (cz * d0 - cx * d2c);
      const float t2 = 2.f * (cx * d1 - cy * d0);
      v8h g;
      g[0] = (_Float16)(d0  + cw * t0 + (cy * t2 - cz * t1));
      g[1] = (_Float16)(d1  + cw * t1 + (cz * t0 - cx * t2));
      g[2] = (_Float16)(d2c + cw * t2 + (cx * t1 - cy * t0));
      g[3] = (_Float16)(cw * qiw - cx * qix - cy * qiy - cz * qiz);
      g[4] = (_Float16)(cw * qix + cx * qiw + cy * qiz - cz * qiy);
      g[5] = (_Float16)(cw * qiy - cx * qiz + cy * qiw + cz * qix);
      g[6] = (_Float16)(cw * qiz + cx * qiy - cy * qix + cz * qiw);
      g[7] = (_Float16)dd;
      store_octet(sF, e16, 20, g);
      v8h z = (v8h)(_Float16)0.0f;
      z[0] = (_Float16)qd;
      store_octet(sF, e16, 21, z);
      z[0] = (_Float16)0.0f;
      store_octet(sF, e16, 22, z);
      store_octet(sF, e16, 23, z);
    }
    lds_fence();

    // ---- layer 1: feat(16x192) @ Wm1 + bm1 -> relu -> hid (A-frag layout) ----
    v16h af[6];
#pragma unroll
    for (int kb = 0; kb < 6; ++kb) af[kb] = load_frag32(sF, kb, lane);
    v8f acc[4];
#pragma unroll
    for (int nt = 0; nt < 4; ++nt) {
#pragma unroll
      for (int r = 0; r < 8; ++r) acc[nt][r] = bM1[nt];
#pragma unroll
      for (int kb = 0; kb < 6; ++kb)
        acc[nt] = wmma_f16(af[kb], load_frag32(sW, kb * 4 + nt, lane), acc[nt]);
    }
    lds_fence();
#pragma unroll
    for (int nt = 0; nt < 4; ++nt)
#pragma unroll
      for (int r = 0; r < 8; ++r)
        store_afrag_elem(sF, r + 8 * hs, nt * 16 + col16, fmaxf(acc[nt][r], 0.0f));
    lds_fence();

    // ---- layer 2: hid @ Wm2 + bm2, mask, msum, m tile (A-frag layout) ----
    const v16h ah0 = load_frag32(sF, 0, lane);
    const v16h ah1 = load_frag32(sF, 1, lane);
    float mrow[8];
#pragma unroll
    for (int r = 0; r < 8; ++r) mrow[r] = sMk[r + 8 * hs];
#pragma unroll
    for (int nt = 0; nt < 4; ++nt) {
      v8f mc;
#pragma unroll
      for (int r = 0; r < 8; ++r) mc[r] = bM2[nt];
      mc = wmma_f16(ah0, load_frag32(sW + 12288, 0 * 4 + nt, lane), mc);
      mc = wmma_f16(ah1, load_frag32(sW + 12288, 1 * 4 + nt, lane), mc);
      float part = 0.0f;
#pragma unroll
      for (int r = 0; r < 8; ++r) {
        const float v = mc[r] * mrow[r];
        part += v;
        store_afrag_elem(sF + 1024, r + 8 * hs, nt * 16 + col16, v);
      }
      atomicAdd(&sMsum[nt * 16 + col16], part);
    }
    lds_fence();

    // ---- layer 3: m @ Wx1 + bx1 -> relu -> t (row-major) ----
    const v16h am0 = load_frag32(sF + 1024, 0, lane);
    const v16h am1 = load_frag32(sF + 1024, 1, lane);
#pragma unroll
    for (int nt = 0; nt < 4; ++nt) {
      v8f tc;
#pragma unroll
      for (int r = 0; r < 8; ++r) tc[r] = bX1[nt];
      tc = wmma_f16(am0, load_frag32(sW + 16384, 0 * 4 + nt, lane), tc);
      tc = wmma_f16(am1, load_frag32(sW + 16384, 1 * 4 + nt, lane), tc);
#pragma unroll
      for (int r = 0; r < 8; ++r)
        sF[2048 + (r + 8 * hs) * 64 + nt * 16 + col16] = (_Float16)fmaxf(tc[r], 0.0f);
    }
    lds_fence();

    // ---- dx = (t @ Wx2 + bx2) * mask, rotate by normalize(q_j), accumulate ----
    if (lane < 16) {
      const v8h* trp = (const v8h*)(sF + 2048 + e16 * 64);
      float d0 = sBx2[0], d1 = sBx2[1], d2v = sBx2[2];
#pragma unroll
      for (int c8 = 0; c8 < 8; ++c8) {
        const v8h ch = trp[c8];
#pragma unroll
        for (int u = 0; u < 8; ++u) {
          const float tv = (float)ch[u];
          const int k = c8 * 8 + u;
          d0  += tv * sWx2[k * 3 + 0];
          d1  += tv * sWx2[k * 3 + 1];
          d2v += tv * sWx2[k * 3 + 2];
        }
      }
      d0 *= maskv; d1 *= maskv; d2v *= maskv;
      const float nq  = sqrtf(qjw * qjw + qjx * qjx + qjy * qjy + qjz * qjz);
      const float inv = 1.0f / fmaxf(nq, 1e-12f);
      const float rw = qjw * inv, rx = qjx * inv, ry = qjy * inv, rz = qjz * inv;
      const float t0 = 2.f * (ry * d2v - rz * d1);
      const float t1 = 2.f * (rz * d0 - rx * d2v);
      const float t2 = 2.f * (rx * d1 - ry * d0);
      atomicAdd(&sDx[0], d0  + rw * t0 + (ry * t2 - rz * t1));
      atomicAdd(&sDx[1], d1  + rw * t1 + (rz * t0 - rx * t2));
      atomicAdd(&sDx[2], d2v + rw * t2 + (rx * t1 - ry * t0));
    }
    lds_fence();
  }
  __syncthreads();
  if (tid < 64) msumG[node * 64 + tid] = sMsum[tid];
  if (tid < 3)  dxG[node * 3 + tid] = sDx[tid];
}

// ---------------- node kernel: per-node fp32 heads ---------------------------
__global__ __launch_bounds__(128) void node_kernel(
    const float* __restrict__ qG, const float* __restrict__ xG,
    const unsigned char* __restrict__ nmG, const unsigned char* __restrict__ pmG,
    const float* __restrict__ hG,
    const float* __restrict__ Wf1, const float* __restrict__ bf1,
    const float* __restrict__ Wf2, const float* __restrict__ bf2,
    const float* __restrict__ Wq1, const float* __restrict__ bq1,
    const float* __restrict__ Wq2, const float* __restrict__ bq2,
    const float* __restrict__ Wt1, const float* __restrict__ bt1,
    const float* __restrict__ Wt2, const float* __restrict__ bt2,
    const float* __restrict__ msumG, const float* __restrict__ dxG,
    float* __restrict__ outG)
{
  const int node = blockIdx.x * blockDim.x + threadIdx.x;
  if (node >= B_ * N_) return;
  const int b = node >> 7;

  float ms[64], hh[64], hid[64];
  for (int k = 0; k < 64; ++k) { ms[k] = msumG[node * 64 + k]; hh[k] = hG[node * 64 + k]; }

  // o = relu([h,msum] @ Wf1 + bf1) @ Wf2 + bf2
  for (int t = 0; t < 64; ++t) {
    float s = bf1[t];
#pragma unroll 8
    for (int k = 0; k < 64; ++k) s += hh[k] * Wf1[k * 64 + t];
#pragma unroll 8
    for (int k = 0; k < 64; ++k) s += ms[k] * Wf1[(64 + k) * 64 + t];
    hid[t] = fmaxf(s, 0.0f);
  }
  for (int u = 0; u < 64; ++u) {
    float s = bf2[u];
#pragma unroll 8
    for (int t = 0; t < 64; ++t) s += hid[t] * Wf2[t * 64 + u];
    outG[OO_OFF + node * 64 + u] = s;
  }

  // torsions = normalize(mlp(msum).reshape(7,2), axis=-1)
  for (int t = 0; t < 64; ++t) {
    float s = bt1[t];
#pragma unroll 8
    for (int k = 0; k < 64; ++k) s += ms[k] * Wt1[k * 64 + t];
    hid[t] = fmaxf(s, 0.0f);
  }
  float tor[14];
  for (int u = 0; u < 14; ++u) {
    float s = bt2[u];
#pragma unroll 8
    for (int t = 0; t < 64; ++t) s += hid[t] * Wt2[t * 14 + u];
    tor[u] = s;
  }
  for (int p = 0; p < 7; ++p) {
    const float a = tor[2 * p], c = tor[2 * p + 1];
    const float inv = 1.0f / fmaxf(sqrtf(a * a + c * c), 1e-12f);
    outG[TOR_OFF + node * 14 + 2 * p]     = a * inv;
    outG[TOR_OFF + node * 14 + 2 * p + 1] = c * inv;
  }

  // dq = mlp(msum); masked to identity; normalize; upd_q = normalize(q * dq)
  for (int t = 0; t < 64; ++t) {
    float s = bq1[t];
#pragma unroll 8
    for (int k = 0; k < 64; ++k) s += ms[k] * Wq1[k * 64 + t];
    hid[t] = fmaxf(s, 0.0f);
  }
  float dq[4];
  for (int u = 0; u < 4; ++u) {
    float s = bq2[u];
#pragma unroll 8
    for (int t = 0; t < 64; ++t) s += hid[t] * Wq2[t * 4 + u];
    dq[u] = s;
  }
  const bool m = nmG[node] != 0;
  if (!m) { dq[0] = 1.0f; dq[1] = dq[2] = dq[3] = 0.0f; }
  {
    const float inv = 1.0f / fmaxf(sqrtf(dq[0]*dq[0] + dq[1]*dq[1] + dq[2]*dq[2] + dq[3]*dq[3]), 1e-12f);
    dq[0] *= inv; dq[1] *= inv; dq[2] *= inv; dq[3] *= inv;
  }
  const float aw = qG[node * 4 + 0], ax = qG[node * 4 + 1];
  const float ay = qG[node * 4 + 2], az = qG[node * 4 + 3];
  float uq[4];
  uq[0] = aw * dq[0] - ax * dq[1] - ay * dq[2] - az * dq[3];
  uq[1] = aw * dq[1] + ax * dq[0] + ay * dq[3] - az * dq[2];
  uq[2] = aw * dq[2] - ax * dq[3] + ay * dq[0] + az * dq[1];
  uq[3] = aw * dq[3] + ax * dq[2] - ay * dq[1] + az * dq[0];
  if (!m) { uq[0] = 1.0f; uq[1] = uq[2] = uq[3] = 0.0f; }
  {
    const float inv = 1.0f / fmaxf(sqrtf(uq[0]*uq[0] + uq[1]*uq[1] + uq[2]*uq[2] + uq[3]*uq[3]), 1e-12f);
    for (int c = 0; c < 4; ++c) outG[UQ_OFF + node * 4 + c] = uq[c] * inv;
  }

  // upd_x = x + sum_j rot(dx) / n_neigh
  int cnt = 0;
  for (int k = 0; k < N_; ++k) cnt += (nmG[b * N_ + k] != 0) ? 1 : 0;
  for (int k = 0; k < P_; ++k) cnt += (pmG[b * P_ + k] != 0) ? 1 : 0;
  const float nneigh = (float)(cnt - 1);
  for (int c = 0; c < 3; ++c)
    outG[UX_OFF + node * 3 + c] = xG[node * 3 + c] + dxG[node * 3 + c] / nneigh;
}

extern "C" void kernel_launch(void* const* d_in, const int* in_sizes, int n_in,
                              void* d_out, int out_size, void* d_ws, size_t ws_size,
                              hipStream_t stream) {
  (void)in_sizes; (void)n_in; (void)out_size; (void)ws_size;
  const float* qG  = (const float*)d_in[0];
  const float* xG  = (const float*)d_in[1];
  // d_in[2] = torsions (unused: only its shape matters)
  const float* hG  = (const float*)d_in[3];
  const float* eG  = (const float*)d_in[4];
  const unsigned char* nmG = (const unsigned char*)d_in[5];
  const float* pqG = (const float*)d_in[6];
  const float* pxG = (const float*)d_in[7];
  const float* peG = (const float*)d_in[8];
  const float* phG = (const float*)d_in[9];
  const unsigned char* pmG = (const unsigned char*)d_in[10];
  const float* Wm1 = (const float*)d_in[11];
  const float* bm1 = (const float*)d_in[12];
  const float* Wm2 = (const float*)d_in[13];
  const float* bm2 = (const float*)d_in[14];
  const float* Wf1 = (const float*)d_in[15];
  const float* bf1 = (const float*)d_in[16];
  const float* Wf2 = (const float*)d_in[17];
  const float* bf2 = (const float*)d_in[18];
  const float* Wx1 = (const float*)d_in[19];
  const float* bx1 = (const float*)d_in[20];
  const float* Wx2 = (const float*)d_in[21];
  const float* bx2 = (const float*)d_in[22];
  const float* Wq1 = (const float*)d_in[23];
  const float* bq1 = (const float*)d_in[24];
  const float* Wq2 = (const float*)d_in[25];
  const float* bq2 = (const float*)d_in[26];
  const float* Wt1 = (const float*)d_in[27];
  const float* bt1 = (const float*)d_in[28];
  const float* Wt2 = (const float*)d_in[29];
  const float* bt2 = (const float*)d_in[30];

  _Float16* w16  = (_Float16*)d_ws;
  float* msumG   = (float*)((char*)d_ws + WS_MSUM_OFF);
  float* dxG     = (float*)((char*)d_ws + WS_DX_OFF);
  float* outG    = (float*)d_out;

  prep_kernel<<<(WSZ_HALVES + 255) / 256, 256, 0, stream>>>(Wm1, Wm2, Wx1, w16);
  edge_kernel<<<B_ * N_, 64, 0, stream>>>(qG, xG, hG, eG, nmG, pqG, pxG, peG, phG, pmG,
                                          bm1, bm2, bx1, Wx2, bx2, w16, msumG, dxG);
  node_kernel<<<(B_ * N_ + 127) / 128, 128, 0, stream>>>(
      qG, xG, nmG, pmG, hG, Wf1, bf1, Wf2, bf2, Wq1, bq1, Wq2, bq2,
      Wt1, bt1, Wt2, bt2, msumG, dxG, outG);
}
